// TernaryTransformerEncoderLayer_75720273428530
// MI455X (gfx1250) — matrix operand order repaired and checked by reading, not compiled, not run on hardware
//
#include <hip/hip_runtime.h>
#include <hip/hip_bf16.h>
#include <math.h>

#define D_MODEL 2048
#define NUM_HEADS 16
#define HEAD_DIM 128
#define D_FF 8192
#define BATCH 2
#define SEQ 2048
#define MTOK (BATCH * SEQ) /* 4096 tokens */

typedef __attribute__((ext_vector_type(16))) __bf16 v16bf;
typedef __attribute__((ext_vector_type(8)))  __bf16 v8bf;
typedef __attribute__((ext_vector_type(8)))  float  v8f;
typedef unsigned short u16t;
typedef unsigned int   u32t;

typedef unsigned int tdm_v4u __attribute__((ext_vector_type(4)));
typedef int          tdm_v4i __attribute__((ext_vector_type(4)));
typedef int          tdm_v8i __attribute__((ext_vector_type(8)));

// ---------- bf16 helpers ----------
static __device__ __forceinline__ u16t f2bf(float f) {
  u32t u = __float_as_uint(f);
  u32t r = (u + 0x7FFFu + ((u >> 16) & 1u)) >> 16; // RNE
  return (u16t)r;
}
// Build a 16-element bf16 fragment from two contiguous 8-element (16B) chunks.
static __device__ __forceinline__ v16bf frag16(const u16t* p0, const u16t* p1) {
  v8bf lo = *(const v8bf*)p0;
  v8bf hi = *(const v8bf*)p1;
  v16bf r;
#pragma unroll
  for (int i = 0; i < 8; ++i) { r[i] = lo[i]; r[i + 8] = hi[i]; }
  return r;
}

// ---------- Tensor Data Mover: DMA a 128-row x 32-col bf16 tile into LDS ----------
// D# packing per CDNA5 ISA ch.8: group0 = {count/flags, lds_addr, global_addr, type},
// group1 = {data_size, tensor dims, tile dims, dim0 stride}. 2D tensor -> groups 2/3 zero.
static __device__ __forceinline__ void tdm_load_tile(u32t lds_byte_addr,
                                                     const u16t* gptr,
                                                     int k_stride_elems) {
  unsigned long long ga = (unsigned long long)(size_t)gptr;
  tdm_v4u g0;
  g0[0] = 1u;                                                  // count=1, user mode
  g0[1] = lds_byte_addr;                                       // lds_addr
  g0[2] = (unsigned)(ga & 0xFFFFFFFFull);                      // global_addr[31:0]
  g0[3] = (unsigned)((ga >> 32) & 0x1FFFFFFull) | (2u << 30);  // addr[56:32] | type=2
  tdm_v8i g1;
  g1[0] = (int)(1u << 16);          // data_size = 1 -> 2 bytes/elem
  g1[1] = (int)(32u << 16);         // tensor_dim0 = 32 (exactly the tile, no clip)
  g1[2] = (int)(128u << 16);        // tensor_dim0_hi=0 | tensor_dim1 = 128
  g1[3] = (int)(32u << 16);         // tensor_dim1_hi=0 | tile_dim0 = 32
  g1[4] = 128;                      // tile_dim1 = 128, tile_dim2 = 0
  g1[5] = k_stride_elems;           // tensor_dim0_stride[31:0] = row pitch (elems)
  g1[6] = 0;                        // stride hi / dim1_stride
  g1[7] = 0;
  tdm_v4i z4 = {0, 0, 0, 0};
#if __clang_major__ >= 23
  tdm_v8i z8 = {0, 0, 0, 0, 0, 0, 0, 0};
  __builtin_amdgcn_tensor_load_to_lds(g0, g1, z4, z4, z8, 0);
#else
  __builtin_amdgcn_tensor_load_to_lds(g0, g1, z4, z4, 0);
#endif
}

static __device__ __forceinline__ u32t lds_off(const void* p) {
  return (u32t)(size_t)p; // low 32 bits of a shared pointer = LDS byte offset
}

// ---------- ternarization ----------
__global__ void zero_floats(float* p, int n) {
  int i = blockIdx.x * blockDim.x + threadIdx.x;
  if (i < n) p[i] = 0.f;
}

__global__ __launch_bounds__(256) void absmean_partial(const float* __restrict__ w, int n,
                                                       float* __restrict__ accum) {
  __shared__ float red[256];
  float s = 0.f;
  for (int i = blockIdx.x * 256 + threadIdx.x; i < n; i += gridDim.x * 256)
    s += fabsf(w[i]);
  red[threadIdx.x] = s;
  __syncthreads();
  for (int off = 128; off > 0; off >>= 1) {
    if (threadIdx.x < off) red[threadIdx.x] += red[threadIdx.x + off];
    __syncthreads();
  }
  if (threadIdx.x == 0) atomicAdd(accum, red[0]);
}

__global__ __launch_bounds__(256) void ternarize_bf16(const float* __restrict__ w,
                                                      const float* __restrict__ accum,
                                                      float inv_n,
                                                      u16t* __restrict__ out, int n) {
  int i = blockIdx.x * 256 + threadIdx.x;
  if (i >= n) return;
  float thr = accum[0] * inv_n;
  float v = w[i];
  float t = (fabsf(v) > thr) ? (v > 0.f ? 1.f : -1.f) : 0.f;
  out[i] = f2bf(t);
}

__global__ __launch_bounds__(256) void cvt_f32_bf16(const float* __restrict__ in,
                                                    u16t* __restrict__ out, int n) {
  int i = blockIdx.x * 256 + threadIdx.x;
  if (i < n) out[i] = f2bf(in[i]);
}

// ---------- GEMM: C[M,N] = A[M,K](bf16,row) x W[N,K](bf16,row)^T ----------
// Block tile 128x128, 8 waves (2x4), wave tile 64x32 = 4x2 WMMA accumulators, K step 32.
// Tiles staged by the Tensor Data Mover, double-buffered in LDS; DMA for step i+1
// overlaps WMMA on step i. One barrier per K step.
// EPI: 0 = f32 out, 1 = bf16 out, 2 = relu + bf16 out
template <int EPI>
__global__ __launch_bounds__(256) void gemm_tn_wmma(const u16t* __restrict__ A,
                                                    const u16t* __restrict__ W,
                                                    float* __restrict__ Cf,
                                                    u16t* __restrict__ Cb,
                                                    int M, int N, int K) {
  __shared__ __align__(16) u16t lA[2][128 * 32];
  __shared__ __align__(16) u16t lB[2][128 * 32];
  const int tid = threadIdx.x;
  const int lane = tid & 31;
  const int wv = tid >> 5;       // 0..7
  const int lanelo = lane & 15;
  const int hi = lane >> 4;      // 0/1
  const int bm = blockIdx.y * 128;
  const int bn = blockIdx.x * 128;
  const int wm = (wv >> 2) * 64; // 0 / 64
  const int wn = (wv & 3) * 32;  // 0 / 32 / 64 / 96

  const u16t* Atile = A + (size_t)bm * K;
  const u16t* Wtile = W + (size_t)bn * K;

  v8f acc[4][2];
#pragma unroll
  for (int mi = 0; mi < 4; ++mi)
#pragma unroll
    for (int ni = 0; ni < 2; ++ni)
#pragma unroll
      for (int g = 0; g < 8; ++g) acc[mi][ni][g] = 0.f;

  if (wv == 0) {                        // one wave drives the TDM
    tdm_load_tile(lds_off(&lA[0][0]), Atile, K);
    tdm_load_tile(lds_off(&lB[0][0]), Wtile, K);
  }

  int buf = 0;
  for (int kk = 0; kk < K; kk += 32) {
    if (wv == 0) __builtin_amdgcn_s_wait_tensorcnt(0); // current buffer landed
    __syncthreads();                                   // publish to all waves
    if (wv == 0 && (kk + 32) < K) {                    // prefetch next tile pair
      tdm_load_tile(lds_off(&lA[buf ^ 1][0]), Atile + kk + 32, K);
      tdm_load_tile(lds_off(&lB[buf ^ 1][0]), Wtile + kk + 32, K);
    }

    const u16t* la = &lA[buf][0];
    const u16t* lb = &lB[buf][0];
    v16bf af[4];
#pragma unroll
    for (int mi = 0; mi < 4; ++mi) {
      const u16t* p = la + (wm + mi * 16 + lanelo) * 32 + hi * 8;
      af[mi] = frag16(p, p + 16);   // A layout: K=hi*8..+7 then +16..+23
    }
    v16bf bfr[2];
#pragma unroll
    for (int ni = 0; ni < 2; ++ni) {
      const u16t* p = lb + (wn + ni * 16 + lanelo) * 32 + hi * 8;
      bfr[ni] = frag16(p, p + 16);  // B layout: K=hi*8..+7 then +16..+23
    }
#pragma unroll
    for (int mi = 0; mi < 4; ++mi)
#pragma unroll
      for (int ni = 0; ni < 2; ++ni)
        acc[mi][ni] = __builtin_amdgcn_wmma_f32_16x16x32_bf16(
            false, af[mi], false, bfr[ni], (short)0, acc[mi][ni], false, false);
    buf ^= 1;
  }

#pragma unroll
  for (int mi = 0; mi < 4; ++mi)
#pragma unroll
    for (int ni = 0; ni < 2; ++ni)
#pragma unroll
      for (int g = 0; g < 8; ++g) {
        int row = bm + wm + mi * 16 + g + 8 * hi;
        int col = bn + wn + ni * 16 + lanelo;
        float v = acc[mi][ni][g];
        if (EPI == 0) {
          Cf[(size_t)row * N + col] = v;
        } else {
          if (EPI == 2) v = fmaxf(v, 0.f);
          Cb[(size_t)row * N + col] = f2bf(v);
        }
      }
}

// ---------- Flash attention (per block: 128 q rows of one (b,h); 8 waves x 16 rows) ----------
__global__ __launch_bounds__(256) void flash_attn_wmma(const u16t* __restrict__ Q,
                                                       const u16t* __restrict__ Km,
                                                       const u16t* __restrict__ V,
                                                       u16t* __restrict__ O) {
  __shared__ __align__(16) u16t lVt[128 * 40];    // V^T tile: [d][k], stride 40 (16B aligned)
  __shared__ __align__(16) u16t lP[8 * 16 * 32];  // per-wave P staging
  const int tid = threadIdx.x;
  const int lane = tid & 31;
  const int wv = tid >> 5;
  const int lanelo = lane & 15;
  const int hi = lane >> 4;
  const int b = blockIdx.z;
  const int h = blockIdx.y;
  const int tok0 = b * SEQ;
  const int q0 = blockIdx.x * 128 + wv * 16;
  const int hoff = h * HEAD_DIM;
  const float scale = 0.08838834764831845f; // 1/sqrt(128)

  // hoist Q fragments (16 x 128 = 4 K-steps of 32) straight from global
  v16bf qf[4];
  {
    const u16t* qp = Q + (size_t)(tok0 + q0 + lanelo) * D_MODEL + hoff + hi * 8;
#pragma unroll
    for (int ks = 0; ks < 4; ++ks) {
      const u16t* p = qp + ks * 32;
      qf[ks] = frag16(p, p + 16);
    }
  }

  v8f o[8];
  float mrow[8], lrow[8];
#pragma unroll
  for (int i = 0; i < 8; ++i) {
#pragma unroll
    for (int g = 0; g < 8; ++g) o[i][g] = 0.f;
    mrow[i] = -1e30f;
    lrow[i] = 0.f;
  }

  for (int kb = 0; kb < SEQ; kb += 32) {
    __syncthreads(); // previous iteration done with lVt
    // stage V block (32 keys x 128 d) transposed into LDS
#pragma unroll
    for (int r = 0; r < 2; ++r) {
      int seg = tid + r * 256;   // 0..511
      int vr = seg >> 4;         // key 0..31
      int c8 = (seg & 15) * 8;   // d col base
      uint4 t = *(const uint4*)&V[(size_t)(tok0 + kb + vr) * D_MODEL + hoff + c8];
      const u16t* e = (const u16t*)&t;
#pragma unroll
      for (int j = 0; j < 8; ++j) lVt[(c8 + j) * 40 + vr] = e[j];
    }
    __syncthreads();

    // S = Q K^T (16 x 32 keys), K fragments straight from global (contiguous d per lane)
    v8f s[2];
#pragma unroll
    for (int ni = 0; ni < 2; ++ni)
#pragma unroll
      for (int g = 0; g < 8; ++g) s[ni][g] = 0.f;

#pragma unroll
    for (int ks = 0; ks < 4; ++ks)
#pragma unroll
      for (int ni = 0; ni < 2; ++ni) {
        const u16t* kp = Km + (size_t)(tok0 + kb + ni * 16 + lanelo) * D_MODEL +
                         hoff + ks * 32 + hi * 8;
        v16bf kf = frag16(kp, kp + 16);
        s[ni] = __builtin_amdgcn_wmma_f32_16x16x32_bf16(
            false, qf[ks], false, kf, (short)0, s[ni], false, false);
      }

    // online softmax: row r = g + 8*hi, 16 cols live across the lanes of the half
    u16t* pw = &lP[wv * 16 * 32];
#pragma unroll
    for (int g = 0; g < 8; ++g) {
      float t0 = s[0][g] * scale;
      float t1 = s[1][g] * scale;
      float mc = fmaxf(t0, t1);
#pragma unroll
      for (int msk = 1; msk < 16; msk <<= 1) mc = fmaxf(mc, __shfl_xor(mc, msk, 32));
      float mn = fmaxf(mrow[g], mc);
      float alpha = __expf(mrow[g] - mn);
      float p0 = __expf(t0 - mn);
      float p1 = __expf(t1 - mn);
      float rs = p0 + p1;
#pragma unroll
      for (int msk = 1; msk < 16; msk <<= 1) rs += __shfl_xor(rs, msk, 32);
      lrow[g] = lrow[g] * alpha + rs;
      mrow[g] = mn;
#pragma unroll
      for (int di = 0; di < 8; ++di) o[di][g] *= alpha;
      pw[(g + 8 * hi) * 32 + lanelo] = f2bf(p0);        // C-layout -> LDS
      pw[(g + 8 * hi) * 32 + 16 + lanelo] = f2bf(p1);
    }

    // O += P V  (P via LDS round-trip into A layout; V^T rows are B layout)
    {
      const u16t* pp = pw + lanelo * 32 + hi * 8;
      v16bf pf = frag16(pp, pp + 16);
#pragma unroll
      for (int di = 0; di < 8; ++di) {
        const u16t* vp = &lVt[(di * 16 + lanelo) * 40 + hi * 8];
        v16bf vf = frag16(vp, vp + 16);
        o[di] = __builtin_amdgcn_wmma_f32_16x16x32_bf16(
            false, pf, false, vf, (short)0, o[di], false, false);
      }
    }
  }

#pragma unroll
  for (int di = 0; di < 8; ++di)
#pragma unroll
    for (int g = 0; g < 8; ++g) {
      float v = o[di][g] / lrow[g];
      O[(size_t)(tok0 + q0 + g + 8 * hi) * D_MODEL + hoff + di * 16 + lanelo] = f2bf(v);
    }
}

// ---------- residual + layernorm (one row per block) ----------
template <bool WF32, bool WBF>
__global__ __launch_bounds__(256) void residual_layernorm(const float* __restrict__ A,
                                                          const float* __restrict__ Bv,
                                                          const float* __restrict__ gamma,
                                                          const float* __restrict__ beta,
                                                          float* __restrict__ Yf,
                                                          u16t* __restrict__ Yb) {
  __shared__ float red[256];
  const int row = blockIdx.x;
  const int tid = threadIdx.x;
  const size_t base = (size_t)row * D_MODEL;
  float vals[8];
  float s = 0.f;
#pragma unroll
  for (int i = 0; i < 8; ++i) {
    float v = A[base + tid + i * 256] + Bv[base + tid + i * 256];
    vals[i] = v;
    s += v;
  }
  red[tid] = s; __syncthreads();
  for (int off = 128; off > 0; off >>= 1) {
    if (tid < off) red[tid] += red[tid + off];
    __syncthreads();
  }
  float mu = red[0] * (1.f / D_MODEL);
  __syncthreads();
  float s2 = 0.f;
#pragma unroll
  for (int i = 0; i < 8; ++i) { float d = vals[i] - mu; s2 += d * d; }
  red[tid] = s2; __syncthreads();
  for (int off = 128; off > 0; off >>= 1) {
    if (tid < off) red[tid] += red[tid + off];
    __syncthreads();
  }
  float rs = rsqrtf(red[0] * (1.f / D_MODEL) + 1e-5f);
#pragma unroll
  for (int i = 0; i < 8; ++i) {
    int c = tid + i * 256;
    float y = (vals[i] - mu) * rs * gamma[c] + beta[c];
    if (WF32) Yf[base + c] = y;
    if (WBF) Yb[base + c] = f2bf(y);
  }
}

// ---------- host launcher ----------
extern "C" void kernel_launch(void* const* d_in, const int* in_sizes, int n_in,
                              void* d_out, int out_size, void* d_ws, size_t ws_size,
                              hipStream_t stream) {
  (void)in_sizes; (void)n_in; (void)out_size; (void)ws_size;
  const float* x  = (const float*)d_in[0];
  const float* Wq = (const float*)d_in[1];
  const float* Wk = (const float*)d_in[2];
  const float* Wv = (const float*)d_in[3];
  const float* Wo = (const float*)d_in[4];
  const float* W1 = (const float*)d_in[5];
  const float* W2 = (const float*)d_in[6];
  const float* g1 = (const float*)d_in[7];
  const float* b1 = (const float*)d_in[8];
  const float* g2 = (const float*)d_in[9];
  const float* b2 = (const float*)d_in[10];

  char* ws = (char*)d_ws;
  size_t off = 0;
  auto alloc = [&](size_t bytes) -> void* {
    void* p = ws + off;
    off += (bytes + 255) & ~(size_t)255;
    return p;
  };

  float* accum = (float*)alloc(6 * sizeof(float));
  u16t* tWq = (u16t*)alloc((size_t)D_MODEL * D_MODEL * 2);
  u16t* tWk = (u16t*)alloc((size_t)D_MODEL * D_MODEL * 2);
  u16t* tWv = (u16t*)alloc((size_t)D_MODEL * D_MODEL * 2);
  u16t* tWo = (u16t*)alloc((size_t)D_MODEL * D_MODEL * 2);
  u16t* tW1 = (u16t*)alloc((size_t)D_FF * D_MODEL * 2);
  u16t* tW2 = (u16t*)alloc((size_t)D_MODEL * D_FF * 2);
  u16t* xbf  = (u16t*)alloc((size_t)MTOK * D_MODEL * 2);  // also reused as ln1 bf16
  u16t* qb   = (u16t*)alloc((size_t)MTOK * D_MODEL * 2);  // q..attn block reused as h1
  u16t* kb   = (u16t*)alloc((size_t)MTOK * D_MODEL * 2);
  u16t* vb   = (u16t*)alloc((size_t)MTOK * D_MODEL * 2);
  u16t* attnb = (u16t*)alloc((size_t)MTOK * D_MODEL * 2);
  float* projf = (float*)alloc((size_t)MTOK * D_MODEL * 4); // o-proj out, later ffn2 out
  float* ln1f  = (float*)alloc((size_t)MTOK * D_MODEL * 4);
  u16t* ln1b = xbf;          // alias: x_bf16 dead after QKV GEMMs
  u16t* h1b  = qb;           // alias: q/k/v/attn (4 x 16.78MB = 67.1MB) dead after o-proj

  // 1) ternarize all six weight tensors
  zero_floats<<<1, 64, 0, stream>>>(accum, 6);
  const float* Wsrc[6] = {Wq, Wk, Wv, Wo, W1, W2};
  u16t* Wdst[6] = {tWq, tWk, tWv, tWo, tW1, tW2};
  int   Wcnt[6] = {D_MODEL * D_MODEL, D_MODEL * D_MODEL, D_MODEL * D_MODEL,
                   D_MODEL * D_MODEL, D_FF * D_MODEL, D_MODEL * D_FF};
  for (int i = 0; i < 6; ++i)
    absmean_partial<<<1024, 256, 0, stream>>>(Wsrc[i], Wcnt[i], accum + i);
  for (int i = 0; i < 6; ++i)
    ternarize_bf16<<<(Wcnt[i] + 255) / 256, 256, 0, stream>>>(
        Wsrc[i], accum + i, 1.0f / (float)Wcnt[i], Wdst[i], Wcnt[i]);

  // 2) x -> bf16
  cvt_f32_bf16<<<(MTOK * D_MODEL + 255) / 256, 256, 0, stream>>>(x, xbf, MTOK * D_MODEL);

  // 3) QKV projections
  dim3 gDD(D_MODEL / 128, MTOK / 128);
  gemm_tn_wmma<1><<<gDD, 256, 0, stream>>>(xbf, tWq, nullptr, qb, MTOK, D_MODEL, D_MODEL);
  gemm_tn_wmma<1><<<gDD, 256, 0, stream>>>(xbf, tWk, nullptr, kb, MTOK, D_MODEL, D_MODEL);
  gemm_tn_wmma<1><<<gDD, 256, 0, stream>>>(xbf, tWv, nullptr, vb, MTOK, D_MODEL, D_MODEL);

  // 4) fused flash attention
  flash_attn_wmma<<<dim3(SEQ / 128, NUM_HEADS, BATCH), 256, 0, stream>>>(qb, kb, vb, attnb);

  // 5) output projection + LN1 (residual with x)
  gemm_tn_wmma<0><<<gDD, 256, 0, stream>>>(attnb, tWo, projf, nullptr, MTOK, D_MODEL, D_MODEL);
  residual_layernorm<true, true><<<MTOK, 256, 0, stream>>>(x, projf, g1, b1, ln1f, ln1b);

  // 6) FFN: relu(ln1 W1^T) W2^T, then LN2 -> d_out
  gemm_tn_wmma<2><<<dim3(D_FF / 128, MTOK / 128), 256, 0, stream>>>(
      ln1b, tW1, nullptr, h1b, MTOK, D_FF, D_MODEL);
  gemm_tn_wmma<0><<<gDD, 256, 0, stream>>>(h1b, tW2, projf, nullptr, MTOK, D_MODEL, D_FF);
  residual_layernorm<true, false><<<MTOK, 256, 0, stream>>>(
      ln1f, projf, g2, b2, (float*)d_out, nullptr);
}
